// PKM_2336462209190
// MI455X (gfx1250) — matrix operand order repaired and checked by reading, not compiled, not run on hardware
//
#include <hip/hip_runtime.h>
#include <hip/hip_bf16.h>
#include <math.h>
#include <stdint.h>

// Problem constants (match reference)
#define N_PTS   8192
#define DIN     2048
#define KD      512      // K_DIM
#define HALFD   256
#define MEMN    512
#define HEADS   4
#define KNN     16
#define OUTD    512
#define QCOLS   (HEADS * KD)   // 2048

typedef __bf16 bf16;
typedef __attribute__((ext_vector_type(16))) __bf16 v16bf;
typedef __attribute__((ext_vector_type(8)))  __bf16 v8bf;
typedef __attribute__((ext_vector_type(8)))  float  v8f;
typedef __attribute__((ext_vector_type(4)))  int    v4i;

// ---- CDNA5 async global->LDS copy path (ASYNCcnt), guarded by __has_builtin
#if defined(__has_builtin)
# if __has_builtin(__builtin_amdgcn_global_load_async_to_lds_b128) && \
     __has_builtin(__builtin_amdgcn_s_wait_asynccnt)
#  define HAVE_ASYNC 1
# endif
#endif
#ifndef HAVE_ASYNC
# define HAVE_ASYNC 0
#endif

// Global (AS1) pointer: generic pointers to global memory share the AS1 bit
// pattern on amdgcn, so an integer round-trip is a safe const-stripping cast.
#define AS1P(p) ((__attribute__((address_space(1))) v4i*)(uintptr_t)(p))
// LDS (AS3) pointer: must be a real addrspacecast (aperture-based generic).
#define AS3P(p) ((__attribute__((address_space(3))) v4i*)(p))

__device__ __forceinline__ v16bf cat16(v8bf lo, v8bf hi) {
  v16bf r;
#pragma unroll
  for (int i = 0; i < 8; ++i) { r[i] = lo[i]; r[i + 8] = hi[i]; }
  return r;
}

__device__ __forceinline__ v8f wmma_bf16(v16bf a, v16bf b, v8f c) {
  // D = A(16x32 bf16) * B(32x16 bf16) + C(16x16 f32)
  return __builtin_amdgcn_wmma_f32_16x16x32_bf16(
      false, a, false, b, (short)0, c, false, false);
}

// Register-resident top-K insertion (fully unrolled -> stays in VGPRs)
__device__ __forceinline__ void topk_insert(float (&bs)[KNN], int (&bi)[KNN],
                                            float v, int id) {
  if (v > bs[KNN - 1]) {
    bs[KNN - 1] = v; bi[KNN - 1] = id;
#pragma unroll
    for (int z = KNN - 1; z > 0; --z) {
      if (bs[z] > bs[z - 1]) {
        float tf = bs[z]; bs[z] = bs[z - 1]; bs[z - 1] = tf;
        int ti2 = bi[z]; bi[z] = bi[z - 1]; bi[z - 1] = ti2;
      }
    }
  }
}

// ---------------- Kernel 1a: straight f32 -> bf16 convert ----------------
__global__ void cvt_f32_bf16(const float* __restrict__ src,
                             bf16* __restrict__ dst, int n) {
  int i = blockIdx.x * blockDim.x + threadIdx.x;
  if (i < n) dst[i] = (bf16)src[i];
}

// ---------------- Kernel 1b: Wq (H,D,K) -> bf16 (D, H*K) row-major -------
__global__ void cvt_wq(const float* __restrict__ Wq, bf16* __restrict__ W2) {
  int i = blockIdx.x * blockDim.x + threadIdx.x;
  if (i >= DIN * QCOLS) return;
  int d = i / QCOLS;
  int j = i % QCOLS;            // h*KD + k
  int h = j / KD, k = j % KD;
  W2[i] = (bf16)Wq[((size_t)h * DIN + d) * KD + k];
}

// ---------------- Kernel 2: Q = X @ W + bq (bf16 WMMA, f32 accum) --------
// Block: 256 thr (8 waves), tile 128(M) x 128(N), K step 32.
__global__ __launch_bounds__(256) void qgemm(
    const bf16* __restrict__ X,   // [N_PTS][DIN]
    const bf16* __restrict__ W,   // [DIN][QCOLS]
    const float* __restrict__ bq, // [QCOLS]
    bf16* __restrict__ Q)         // [N_PTS][QCOLS]
{
  __shared__ bf16 As[128][40];    // [m][k], pad to keep 16B alignment
  __shared__ bf16 Bs[128][40];    // transposed: [n][k]

  const int tid  = threadIdx.x;
  const int lane = tid & 31;
  const int wid  = tid >> 5;
  const int waveM = wid & 3;      // 4 waves down M (32 rows each)
  const int waveN = wid >> 2;     // 2 waves across N (64 cols each)
  const int m0 = blockIdx.y * 128;
  const int n0 = blockIdx.x * 128;

  v8f acc[2][4] = {};

  for (int k0 = 0; k0 < DIN; k0 += 32) {
    { // stage A tile: 128x32 bf16 (16B-contiguous chunks)
      int row = tid >> 1;
      int col = (tid & 1) * 16;
      const bf16* gp = X + (size_t)(m0 + row) * DIN + k0 + col;
#if HAVE_ASYNC
      // CDNA5 async copy: global -> LDS without VGPR round trip (ASYNCcnt)
      __builtin_amdgcn_global_load_async_to_lds_b128(
          AS1P(gp), AS3P(&As[row][col]), 0, 0);
      __builtin_amdgcn_global_load_async_to_lds_b128(
          AS1P(gp + 8), AS3P(&As[row][col + 8]), 0, 0);
#else
      const v8bf* gv = (const v8bf*)gp;
      *(v8bf*)&As[row][col]     = gv[0];
      *(v8bf*)&As[row][col + 8] = gv[1];
#endif
    }
    { // stage B tile transposed: 32x128 -> Bs[n][k] (needs in-LDS transpose)
      int k  = tid >> 3;
      int c0 = (tid & 7) * 16;
      const v8bf* gp = (const v8bf*)(W + (size_t)(k0 + k) * QCOLS + n0 + c0);
      v8bf b0 = gp[0], b1 = gp[1];
#pragma unroll
      for (int i = 0; i < 8; ++i) {
        Bs[c0 + i][k]     = b0[i];
        Bs[c0 + 8 + i][k] = b1[i];
      }
    }
#if HAVE_ASYNC
    __builtin_amdgcn_s_wait_asynccnt(0);
#endif
    __syncthreads();

    const int am  = lane & 15;
    const int akb = (lane < 16) ? 0 : 16;   // A: lanes 0-15 K=0..15
    const int bkb = (lane < 16) ? 0 : 16;   // B: lanes 0-15 K=0..15

    v16bf afrag[2];
#pragma unroll
    for (int mi = 0; mi < 2; ++mi) {
      int m = waveM * 32 + mi * 16 + am;
      afrag[mi] = cat16(*(const v8bf*)&As[m][akb],
                        *(const v8bf*)&As[m][akb + 8]);
    }
#pragma unroll
    for (int ni = 0; ni < 4; ++ni) {
      int c = waveN * 64 + ni * 16 + am;
      v16bf bfrag = cat16(*(const v8bf*)&Bs[c][bkb],
                          *(const v8bf*)&Bs[c][bkb + 8]);
#pragma unroll
      for (int mi = 0; mi < 2; ++mi)
        acc[mi][ni] = wmma_bf16(afrag[mi], bfrag, acc[mi][ni]);
    }
    __syncthreads();
  }

  // bias + store bf16 q
#pragma unroll
  for (int ni = 0; ni < 4; ++ni) {
    int col = n0 + waveN * 64 + ni * 16 + (lane & 15);
    float bias = bq[col];
#pragma unroll
    for (int mi = 0; mi < 2; ++mi) {
#pragma unroll
      for (int r = 0; r < 8; ++r) {
        int row = m0 + waveM * 32 + mi * 16 + r + ((lane >> 4) << 3);
        Q[(size_t)row * QCOLS + col] = (bf16)(acc[mi][ni][r] + bias);
      }
    }
  }
}

// ---------------- Kernel 3: fused scores + top-k + softmax + gather ------
// Block handles 16 query rows, loops over 4 heads. 256 threads (8 waves).
__global__ __launch_bounds__(256) void pkm_fused(
    const bf16* __restrict__ Q,    // [N_PTS][QCOLS]
    const bf16* __restrict__ Kb,   // [(h*2+half)*MEMN + m][HALFD]  (bf16)
    const float* __restrict__ V,   // [MEMN*MEMN][OUTD]
    float* __restrict__ out)       // [N_PTS][OUTD]
{
  __shared__ bf16  q_lds[16][520];        // 16 x 512 q (one head), padded
  __shared__ float s_lds[16][MEMN];       // scores, reused for both halves
  __shared__ float t_s[2][16][KNN];
  __shared__ int   t_i[2][16][KNN];
  __shared__ float c_w[16][KNN];
  __shared__ int   c_i[16][KNN];

  const int tid  = threadIdx.x;
  const int lane = tid & 31;
  const int wid  = tid >> 5;
  const int n0   = blockIdx.x * 16;
  const int orow = tid >> 4;   // 0..15 output row
  const int ocg  = tid & 15;   // 0..15 -> 32 output cols

  float oacc[32];
#pragma unroll
  for (int i = 0; i < 32; ++i) oacc[i] = 0.f;

  for (int h = 0; h < HEADS; ++h) {
    { // load q tile for this head: 16 x 512 bf16
      int row = tid >> 4;
      int c0  = (tid & 15) * 32;
      const bf16* gp = Q + (size_t)(n0 + row) * QCOLS + h * KD + c0;
#if HAVE_ASYNC
#pragma unroll
      for (int i = 0; i < 4; ++i)
        __builtin_amdgcn_global_load_async_to_lds_b128(
            AS1P(gp + i * 8), AS3P(&q_lds[row][c0 + i * 8]), 0, 0);
      __builtin_amdgcn_s_wait_asynccnt(0);
#else
      const v8bf* gv = (const v8bf*)gp;
#pragma unroll
      for (int i = 0; i < 4; ++i)
        *(v8bf*)&q_lds[row][c0 + i * 8] = gv[i];
#endif
    }
    __syncthreads();

    for (int half = 0; half < 2; ++half) {
      // ---- scores: s[16][512] = q_half(16x256) x keysT(256x512), WMMA ----
      const int am   = lane & 15;
      const int akb  = (lane < 16) ? 0 : 16;
      const int bofs = (lane < 16) ? 0 : 16;
      v8f acc[4] = {};   // 4 column tiles of 16 per wave (8 waves * 64 = 512)
      for (int kk = 0; kk < 8; ++kk) {
        int kq = half * HALFD + kk * 32;
        v16bf a = cat16(*(const v8bf*)&q_lds[am][kq + akb],
                        *(const v8bf*)&q_lds[am][kq + akb + 8]);
#pragma unroll
        for (int ti = 0; ti < 4; ++ti) {
          int m = (wid * 4 + ti) * 16 + am;
          const bf16* bp =
              Kb + ((size_t)((h * 2 + half) * MEMN + m)) * HALFD + kk * 32 + bofs;
          v16bf b = cat16(*(const v8bf*)bp, *(const v8bf*)(bp + 8));
          acc[ti] = wmma_bf16(a, b, acc[ti]);
        }
      }
#pragma unroll
      for (int ti = 0; ti < 4; ++ti) {
        int col = (wid * 4 + ti) * 16 + am;
#pragma unroll
        for (int r = 0; r < 8; ++r)
          s_lds[r + ((lane >> 4) << 3)][col] = acc[ti][r];
      }
      __syncthreads();

      // ---- top-16 of 512 per row (16 threads, registers) ----
      if (tid < 16) {
        float bs[KNN]; int bi[KNN];
#pragma unroll
        for (int i = 0; i < KNN; ++i) { bs[i] = -INFINITY; bi[i] = 0; }
        for (int m = 0; m < MEMN; ++m)
          topk_insert(bs, bi, s_lds[tid][m], m);
#pragma unroll
        for (int i = 0; i < KNN; ++i) {
          t_s[half][tid][i] = bs[i];
          t_i[half][tid][i] = bi[i];
        }
      }
      __syncthreads();   // also guards s_lds reuse for next half
    }

    // ---- combine 16x16 -> top-16, softmax (16 threads) ----
    if (tid < 16) {
      float bs[KNN]; int bi[KNN];
#pragma unroll
      for (int i = 0; i < KNN; ++i) { bs[i] = -INFINITY; bi[i] = 0; }
      for (int i = 0; i < KNN; ++i) {
        float s1 = t_s[0][tid][i];
        int   i1 = t_i[0][tid][i] * MEMN;
        for (int j = 0; j < KNN; ++j)
          topk_insert(bs, bi, s1 + t_s[1][tid][j], i1 + t_i[1][tid][j]);
      }
      float mx = bs[0];
      float w[KNN];
      float sum = 0.f;
#pragma unroll
      for (int e = 0; e < KNN; ++e) { w[e] = __expf(bs[e] - mx); sum += w[e]; }
      float inv = 1.f / sum;
#pragma unroll
      for (int e = 0; e < KNN; ++e) {
        c_w[tid][e] = w[e] * inv;
        c_i[tid][e] = bi[e];
      }
    }
    __syncthreads();

    // ---- gather values rows, accumulate into registers across heads ----
    for (int e = 0; e < KNN; ++e) {
      float wgt = c_w[orow][e];
      int   idx = c_i[orow][e];
      if (e + 1 < KNN) {
        // next row -> global_prefetch_b8 (HBM latency hiding)
        __builtin_prefetch(
            (const void*)(V + (size_t)c_i[orow][e + 1] * OUTD + ocg * 32), 0, 1);
      }
      const float4* vp = (const float4*)(V + (size_t)idx * OUTD + ocg * 32);
#pragma unroll
      for (int qd = 0; qd < 8; ++qd) {
        float4 vv = vp[qd];
        oacc[qd * 4 + 0] += wgt * vv.x;
        oacc[qd * 4 + 1] += wgt * vv.y;
        oacc[qd * 4 + 2] += wgt * vv.z;
        oacc[qd * 4 + 3] += wgt * vv.w;
      }
    }
    __syncthreads();   // before next head overwrites q_lds / c_w / c_i
  }

  // write output once (no atomics, fully overwrites d_out)
  float4* op = (float4*)(out + (size_t)(n0 + orow) * OUTD + ocg * 32);
#pragma unroll
  for (int qd = 0; qd < 8; ++qd)
    op[qd] = make_float4(oacc[qd * 4 + 0], oacc[qd * 4 + 1],
                         oacc[qd * 4 + 2], oacc[qd * 4 + 3]);
}

// ---------------- host-side launch ----------------
extern "C" void kernel_launch(void* const* d_in, const int* in_sizes, int n_in,
                              void* d_out, int out_size, void* d_ws, size_t ws_size,
                              hipStream_t stream) {
  const float* x      = (const float*)d_in[0];  // (N, D_IN)
  const float* Wq     = (const float*)d_in[1];  // (H, D_IN, K_DIM)
  const float* bq     = (const float*)d_in[2];  // (H, K_DIM)
  const float* keys   = (const float*)d_in[3];  // (H, 2, MEM, HALF)
  const float* values = (const float*)d_in[4];  // (MEM*MEM, OUT)
  float* out = (float*)d_out;

  // workspace layout (bf16): x | W(reordered) | keys | q   (~78 MB total)
  bf16* xb = (bf16*)d_ws;
  bf16* wb = xb + (size_t)N_PTS * DIN;
  bf16* kb = wb + (size_t)DIN * QCOLS;
  bf16* qb = kb + (size_t)HEADS * 2 * MEMN * HALFD;

  const int nx = N_PTS * DIN;
  const int nw = DIN * QCOLS;
  const int nk = HEADS * 2 * MEMN * HALFD;

  cvt_f32_bf16<<<(nx + 255) / 256, 256, 0, stream>>>(x, xb, nx);
  cvt_wq      <<<(nw + 255) / 256, 256, 0, stream>>>(Wq, wb);
  cvt_f32_bf16<<<(nk + 255) / 256, 256, 0, stream>>>(keys, kb, nk);

  dim3 gg(QCOLS / 128, N_PTS / 128);
  qgemm<<<gg, 256, 0, stream>>>(xb, wb, bq, qb);

  pkm_fused<<<N_PTS / 16, 256, 0, stream>>>(qb, kb, values, out);
}
